// DeltaNet_84748294684731
// MI455X (gfx1250) — compile-verified, matrix operand order checked by reading, not run on hardware
//
#include <hip/hip_runtime.h>
#include <hip/hip_bf16.h>

typedef _Float16 v16h __attribute__((ext_vector_type(16)));
typedef _Float16 v8h  __attribute__((ext_vector_type(8)));
typedef float    v8f  __attribute__((ext_vector_type(8)));

#define CB 2
#define CL 4096
#define CD 1024
#define CH 4
#define CDK 256
#define CDV 256
#define NCHUNK 128          // L / 32
#define NTOK (CB*CL)        // 8192
#define NROWS (NTOK*CH)     // 32768
#define KPAD 1056           // 1040 padded to multiple of 32

// Async global->LDS copy path (ASYNCcnt) if the toolchain exposes it.
#if defined(__has_builtin)
#if __has_builtin(__builtin_amdgcn_global_load_async_to_lds_b128) && \
    __has_builtin(__builtin_amdgcn_s_wait_asynccnt)
#define USE_ASYNC_COPY 1
#endif
#endif
#ifndef USE_ASYNC_COPY
#define USE_ASYNC_COPY 0
#endif

#if USE_ASYNC_COPY
// Builtin signature (from compiler diagnostic): param0 is
// 'int __vector(4) __device__ *' i.e. addrspace(1) pointer to vector_size(16).
typedef int v4i_vs __attribute__((vector_size(16)));
typedef __attribute__((address_space(1))) v4i_vs* g_v4i_p;
typedef __attribute__((address_space(3))) v4i_vs* l_v4i_p;
__device__ __forceinline__ void async_copy_b128(const _Float16* g, _Float16* l) {
  // per-lane 16B global -> LDS, tracked by ASYNCcnt
  __builtin_amdgcn_global_load_async_to_lds_b128(
      (g_v4i_p)(uintptr_t)g,
      (l_v4i_p)(unsigned int)(uintptr_t)l,
      0, 0);
}
#endif

// ---------------------------------------------------------------- conversions
__global__ void cvt_f16_kernel(const float* __restrict__ in,
                               _Float16* __restrict__ out, int n) {
  int i = blockIdx.x * blockDim.x + threadIdx.x;
  if (i < n) out[i] = (_Float16)in[i];
}

__global__ void cvt_pad_kernel(const float* __restrict__ in,
                               _Float16* __restrict__ out,
                               int rows, int kin, int kout) {
  int i = blockIdx.x * blockDim.x + threadIdx.x;
  if (i >= rows * kout) return;
  int r = i / kout, c = i % kout;
  out[i] = (c < kin) ? (_Float16)in[(size_t)r * kin + c] : (_Float16)0.f;
}

// ---------------------------------------------------------------- WMMA GEMM
// C[M,N] (f32) = A[M,K] (f16, row-major) * W[N,K]^T (f16, row-major)
// Block tile 128x128, 8 waves, each wave 32(M) x 64(N) = 2x4 WMMA frags.
// Double-buffered LDS tiles; async global->LDS when available.
#define GT_M 128
#define GT_N 128
#define GT_K 32
#define LDH  40   // LDS row stride in halves (80B, 16B aligned, conflict pad)

__device__ __forceinline__ v16h ldsFrag(const _Float16* rowBase, int lane) {
  // 16-bit A/B fragment layout: lanes 0-15 -> row = lane, K segs {0..7,16..23}
  //                             lanes 16-31 -> row = lane-16, K segs {8..15,24..31}
  const int r  = lane & 15;
  const int k0 = (lane & 16) ? 8 : 0;
  const _Float16* p = rowBase + r * LDH + k0;
  v8h lo = *reinterpret_cast<const v8h*>(p);
  v8h hi = *reinterpret_cast<const v8h*>(p + 16);
  v16h f;
#pragma unroll
  for (int e = 0; e < 8; ++e) { f[e] = lo[e]; f[e + 8] = hi[e]; }
  return f;
}

__global__ __launch_bounds__(256)
void gemm_f16f32_kernel(const _Float16* __restrict__ A,
                        const _Float16* __restrict__ W,
                        float* __restrict__ C,
                        int M, int N, int K) {
  __shared__ alignas(16) _Float16 sA[2][GT_M * LDH];
  __shared__ alignas(16) _Float16 sB[2][GT_N * LDH];
  const int tid  = threadIdx.x;
  const int lane = tid & 31;
  const int wv   = tid >> 5;
  const int wm   = (wv >> 1) * 32;   // 0..96
  const int wn   = (wv & 1) * 64;    // 0,64
  const int m0   = blockIdx.y * GT_M;
  const int n0   = blockIdx.x * GT_N;

  v8f acc[2][4];
#pragma unroll
  for (int i = 0; i < 2; ++i)
#pragma unroll
    for (int j = 0; j < 4; ++j)
#pragma unroll
      for (int e = 0; e < 8; ++e) acc[i][j][e] = 0.f;

  // Each thread owns chunks tid and tid+256 of each 128x32 tile (16B chunks).
  auto stage = [&](int buf, int k0) {
#pragma unroll
    for (int it = 0; it < 2; ++it) {
      int q   = tid + it * 256;       // 0..511
      int row = q >> 2;               // 0..127
      int cc  = (q & 3) * 8;          // 0,8,16,24 halves
#if USE_ASYNC_COPY
      async_copy_b128(&A[(size_t)(m0 + row) * K + k0 + cc], &sA[buf][row * LDH + cc]);
      async_copy_b128(&W[(size_t)(n0 + row) * K + k0 + cc], &sB[buf][row * LDH + cc]);
#else
      *reinterpret_cast<v8h*>(&sA[buf][row * LDH + cc]) =
          *reinterpret_cast<const v8h*>(&A[(size_t)(m0 + row) * K + k0 + cc]);
      *reinterpret_cast<v8h*>(&sB[buf][row * LDH + cc]) =
          *reinterpret_cast<const v8h*>(&W[(size_t)(n0 + row) * K + k0 + cc]);
#endif
    }
  };

  stage(0, 0);
  int buf = 0;
  for (int k0 = 0; k0 < K; k0 += GT_K) {
    const bool hasNext = (k0 + GT_K) < K;
    if (hasNext) stage(buf ^ 1, k0 + GT_K);
#if USE_ASYNC_COPY
    if (hasNext) __builtin_amdgcn_s_wait_asynccnt(4);  // current stage landed
    else         __builtin_amdgcn_s_wait_asynccnt(0);
#else
    if (hasNext) {
      __builtin_prefetch(&A[(size_t)(m0 + (tid >> 1)) * K + k0 + GT_K], 0, 1);
      __builtin_prefetch(&W[(size_t)(n0 + (tid >> 1)) * K + k0 + GT_K], 0, 1);
    }
#endif
    __syncthreads();
    v16h af[2], bf[4];
#pragma unroll
    for (int i = 0; i < 2; ++i) af[i] = ldsFrag(&sA[buf][(wm + 16 * i) * LDH], lane);
#pragma unroll
    for (int j = 0; j < 4; ++j) bf[j] = ldsFrag(&sB[buf][(wn + 16 * j) * LDH], lane);
#pragma unroll
    for (int i = 0; i < 2; ++i)
#pragma unroll
      for (int j = 0; j < 4; ++j)
        acc[i][j] = __builtin_amdgcn_wmma_f32_16x16x32_f16(
            false, af[i], false, bf[j], (short)0, acc[i][j], false, false);
    __syncthreads();   // buffer free before stage k0+2*GT_K overwrites it
    buf ^= 1;
  }

  // C/D layout: VGPR v, lanes 0-15 -> M=v,   N=lane
  //                      lanes16-31 -> M=v+8, N=lane-16
  const int crow = (lane & 16) ? 8 : 0;
  const int ccol = lane & 15;
#pragma unroll
  for (int i = 0; i < 2; ++i)
#pragma unroll
    for (int j = 0; j < 4; ++j) {
      size_t base = (size_t)(m0 + wm + 16 * i + crow) * N + (n0 + wn + 16 * j + ccol);
#pragma unroll
      for (int v = 0; v < 8; ++v) C[base + (size_t)v * N] = acc[i][j][v];
    }
}

// ---------------------------------------------------------------- short conv + SiLU
__global__ void shortconv_silu_kernel(const float* __restrict__ x,
                                      const float* __restrict__ w,
                                      float* __restrict__ y, int total) {
  int i = blockIdx.x * blockDim.x + threadIdx.x;
  if (i >= total) return;
  int c = i & (CD - 1);
  int l = (i >> 10) & (CL - 1);
  float acc = 0.f;
#pragma unroll
  for (int j = 0; j < 4; ++j) {
    int ll = l + j - 3;
    if (ll >= 0) acc += x[(size_t)i + (size_t)(j - 3) * CD] * w[c * 4 + j];
  }
  y[i] = acc / (1.f + __expf(-acc));   // silu
}

// ---------------------------------------------------------------- beta
__global__ void beta_kernel(const float* __restrict__ hs,
                            const float* __restrict__ bw,
                            float* __restrict__ beta, int rows) {
  int i = blockIdx.x * blockDim.x + threadIdx.x;
  if (i >= rows) return;
  int h = i & 3, t = i >> 2;
  const float4* a = (const float4*)&hs[(size_t)t * CD];
  const float4* b = (const float4*)&bw[(size_t)h * CD];
  float acc = 0.f;
  for (int d = 0; d < CD / 4; ++d) {
    float4 av = a[d], bv = b[d];
    acc += av.x * bv.x + av.y * bv.y + av.z * bv.z + av.w * bv.w;
  }
  float s = 1.f / (1.f + __expf(-acc));
  beta[i] = fmaxf(s, 1e-6f);
}

// ---------------------------------------------------------------- delta-rule chunk prep
// per block: one (b,h,chunk). l2norm(q),l2norm(k); M=strict_lower(beta_i*k_i.k_j);
// T=(I+M)^-1 (unit lower); u=T@(v*beta); w=T@(k*beta). Writes qn,kn,u,w in (b,h,l,d).
__global__ __launch_bounds__(256)
void deltaprep_kernel(const float* __restrict__ qc, const float* __restrict__ kc,
                      const float* __restrict__ vc, const float* __restrict__ beta,
                      float* __restrict__ qn, float* __restrict__ kn,
                      float* __restrict__ u, float* __restrict__ w) {
  __shared__ alignas(16) float sK[32 * 256];
  __shared__ float sM[1024], sT[1024];
  __shared__ float sRedA[256], sRedB[256];
  __shared__ float sScQ[32], sScK[32], sBeta[32];

  int blk = blockIdx.x;
  int n = blk % NCHUNK, bh = blk / NCHUNK, h = bh & 3, b = bh >> 2;
  int tid = threadIdx.x;
  int i = tid >> 3, s = tid & 7;

  size_t grow = ((size_t)(b * CL) + n * 32 + i) * CD + h * CDK + s * 32;
  float pq = 0.f, pk = 0.f;
#pragma unroll
  for (int e = 0; e < 8; ++e) {
    float4 qv = *(const float4*)&qc[grow + e * 4];
    float4 kv = *(const float4*)&kc[grow + e * 4];
    pq += qv.x * qv.x + qv.y * qv.y + qv.z * qv.z + qv.w * qv.w;
    pk += kv.x * kv.x + kv.y * kv.y + kv.z * kv.z + kv.w * kv.w;
  }
  sRedA[tid] = pq; sRedB[tid] = pk;
  if (tid < 32) sBeta[tid] = beta[((size_t)(b * CL) + n * 32 + tid) * CH + h];
  __syncthreads();
  if (tid < 32) {
    float sq = 0.f, sk = 0.f;
#pragma unroll
    for (int e = 0; e < 8; ++e) { sq += sRedA[tid * 8 + e]; sk += sRedB[tid * 8 + e]; }
    sScQ[tid] = rsqrtf(sq + 1e-6f);
    sScK[tid] = rsqrtf(sk + 1e-6f);
  }
  __syncthreads();
  size_t orow = (((size_t)bh) * CL + n * 32 + i) * CDK + s * 32;
  float scq = sScQ[i], sck = sScK[i];
#pragma unroll
  for (int e = 0; e < 8; ++e) {
    float4 qv = *(const float4*)&qc[grow + e * 4];
    float4 kv = *(const float4*)&kc[grow + e * 4];
    qv.x *= scq; qv.y *= scq; qv.z *= scq; qv.w *= scq;
    kv.x *= sck; kv.y *= sck; kv.z *= sck; kv.w *= sck;
    *(float4*)&qn[orow + e * 4] = qv;
    *(float4*)&kn[orow + e * 4] = kv;
    *(float4*)&sK[i * 256 + s * 32 + e * 4] = kv;
  }
  __syncthreads();
  // M = strict lower of beta_i * (kn_i . kn_j)
#pragma unroll
  for (int e2 = 0; e2 < 4; ++e2) {
    int idx = tid * 4 + e2;
    int mi = idx >> 5, mj = idx & 31;
    float acc = 0.f;
    if (mj < mi) {
      for (int d = 0; d < 256; ++d) acc += sK[mi * 256 + d] * sK[mj * 256 + d];
      acc *= sBeta[mi];
    }
    sM[mi * 32 + mj] = acc;
  }
  __syncthreads();
  // T = (I+M)^-1, column-parallel forward substitution
  if (tid < 32) {
    int j = tid;
    for (int ii = 0; ii < 32; ++ii) {
      float t;
      if (ii < j) t = 0.f;
      else if (ii == j) t = 1.f;
      else {
        t = 0.f;
        for (int p = j; p < ii; ++p) t -= sM[ii * 32 + p] * sT[p * 32 + j];
      }
      sT[ii * 32 + j] = t;
    }
  }
  __syncthreads();
  // u = T @ (v*beta); w = T @ (kn*beta); thread owns column d = tid
  {
    int d = tid;
    size_t gv = ((size_t)(b * CL) + n * 32) * CD + h * CDV + d;
    float accu[32], accw[32];
#pragma unroll
    for (int ii = 0; ii < 32; ++ii) { accu[ii] = 0.f; accw[ii] = 0.f; }
    for (int p = 0; p < 32; ++p) {
      float tb = sBeta[p];
      float vv = vc[gv + (size_t)p * CD];
      float kk = sK[p * 256 + d];
#pragma unroll
      for (int ii = 0; ii < 32; ++ii) {
        if (ii >= p) {
          float tpb = sT[ii * 32 + p] * tb;
          accu[ii] += tpb * vv;
          accw[ii] += tpb * kk;
        }
      }
    }
    size_t obase = (((size_t)bh) * CL + n * 32) * CDK + d;
#pragma unroll
    for (int ii = 0; ii < 32; ++ii) {
      u[obase + (size_t)ii * CDK] = accu[ii];
      w[obase + (size_t)ii * CDK] = accw[ii];
    }
  }
}

// ---------------------------------------------------------------- sequential scan
// block = (b,h, dv-slice of 16). S[256][16] in LDS. 128 chunk steps.
__global__ __launch_bounds__(256)
void scan_kernel(const float* __restrict__ qn, const float* __restrict__ kn,
                 const float* __restrict__ u, const float* __restrict__ w,
                 float* __restrict__ dout) {
  extern __shared__ float smem[];
  float* sS   = smem;            // 256*16
  float* sQ   = sS + 4096;       // 32*256
  float* sK   = sQ + 8192;
  float* sW   = sK + 8192;
  float* sU   = sW + 8192;       // 32*16
  float* sUa  = sU + 512;        // 32*16
  float* sAttn= sUa + 512;       // 32*32

  int blk = blockIdx.x;
  int sl = blk & 15, bh = blk >> 4, h = bh & 3, b = bh >> 2;
  int d0 = sl * 16;
  int tid = threadIdx.x;
  int li = tid >> 3, ls = tid & 7;
#pragma unroll
  for (int e = 0; e < 16; ++e) sS[tid * 16 + e] = 0.f;

  for (int n = 0; n < NCHUNK; ++n) {
    size_t cb = (((size_t)bh) * CL + n * 32) * CDK;
#pragma unroll
    for (int e = 0; e < 8; ++e) {
      int d = ls * 32 + e * 4;
      *(float4*)&sQ[li * 256 + d] = *(const float4*)&qn[cb + (size_t)li * 256 + d];
      *(float4*)&sK[li * 256 + d] = *(const float4*)&kn[cb + (size_t)li * 256 + d];
      *(float4*)&sW[li * 256 + d] = *(const float4*)&w [cb + (size_t)li * 256 + d];
    }
#pragma unroll
    for (int r = 0; r < 2; ++r) {
      int e = tid * 2 + r, ui = e >> 4, ud = e & 15;
      sU[e] = u[cb + (size_t)ui * 256 + d0 + ud];
    }
    if (n + 1 < NCHUNK) {   // pull next chunk toward L2/L0
      size_t nb = (((size_t)bh) * CL + (n + 1) * 32) * CDK + (size_t)tid * 32;
      __builtin_prefetch(&qn[nb], 0, 1);
      __builtin_prefetch(&kn[nb], 0, 1);
      __builtin_prefetch(&w[nb], 0, 1);
    }
    __syncthreads();
    // wS, qS, u_adj
    float qs[2];
#pragma unroll
    for (int r = 0; r < 2; ++r) {
      int e = tid * 2 + r, ri = e >> 4, rd = e & 15;
      float aw = 0.f, aq = 0.f;
      for (int dk = 0; dk < 256; ++dk) {
        float sv = sS[dk * 16 + rd];
        aw += sW[ri * 256 + dk] * sv;
        aq += sQ[ri * 256 + dk] * sv;
      }
      sUa[e] = sU[e] - aw;
      qs[r] = aq;
    }
    // attn = tril(q k^T) incl diag
#pragma unroll
    for (int r = 0; r < 4; ++r) {
      int e = tid * 4 + r, ai = e >> 5, aj = e & 31;
      float acc = 0.f;
      if (aj <= ai)
        for (int dk = 0; dk < 256; ++dk) acc += sQ[ai * 256 + dk] * sK[aj * 256 + dk];
      sAttn[e] = acc;
    }
    __syncthreads();
    // out = qS + attn @ u_adj
#pragma unroll
    for (int r = 0; r < 2; ++r) {
      int e = tid * 2 + r, ri = e >> 4, rd = e & 15;
      float acc = qs[r];
      for (int p = 0; p < 32; ++p) acc += sAttn[ri * 32 + p] * sUa[p * 16 + rd];
      dout[((size_t)(b * CL) + n * 32 + ri) * CD + h * CDV + d0 + rd] = acc;
    }
    // S += k^T @ u_adj   (thread owns dk = tid)
    {
      float kcol[32];
#pragma unroll
      for (int p = 0; p < 32; ++p) kcol[p] = sK[p * 256 + tid];
#pragma unroll
      for (int dd = 0; dd < 16; ++dd) {
        float acc = 0.f;
        for (int p = 0; p < 32; ++p) acc += kcol[p] * sUa[p * 16 + dd];
        sS[tid * 16 + dd] += acc;
      }
    }
    __syncthreads();
  }
}

// ---------------------------------------------------------------- fused FIR (K=5 & K=64)
__global__ __launch_bounds__(256)
void fir_kernel(const float* __restrict__ v, const float* __restrict__ fsw,
                const float* __restrict__ flw, float* __restrict__ os,
                float* __restrict__ ol) {
  extern __shared__ float sX[];  // 127 * 256
  int blk = blockIdx.x;
  int lt = blk & 63, bh = blk >> 6, h = bh & 3, b = bh >> 2;
  int l0 = lt * 64;
  int d = threadIdx.x;
  float fl[64], fs[5];
#pragma unroll
  for (int j = 0; j < 64; ++j) fl[j] = flw[((size_t)(h * 256) + d) * 64 + j];
#pragma unroll
  for (int j = 0; j < 5; ++j) fs[j] = fsw[((size_t)(h * 256) + d) * 5 + j];
  for (int r = 0; r < 127; ++r) {
    int l = l0 - 63 + r;
    sX[r * 256 + d] = (l >= 0) ? v[((size_t)(b * CL) + l) * CD + h * CDV + d] : 0.f;
  }
  __syncthreads();
  for (int r = 0; r < 64; ++r) {
    float accl = 0.f;
#pragma unroll
    for (int j = 0; j < 64; ++j) accl += sX[(r + j) * 256 + d] * fl[j];
    float accs = 0.f;
#pragma unroll
    for (int j = 0; j < 5; ++j) accs += sX[(r + 59 + j) * 256 + d] * fs[j];
    size_t o = ((size_t)(b * CL) + l0 + r) * CD + h * CDV + d;
    ol[o] = accl;
    os[o] = accs;
  }
}

// ---------------------------------------------------------------- stats + gate_in (f16, padded K)
__global__ __launch_bounds__(256)
void stats_kernel(const float* __restrict__ hs, const float* __restrict__ fsrt,
                  const float* __restrict__ flng, const float* __restrict__ dlt,
                  const float* __restrict__ vd, _Float16* __restrict__ gate_in) {
  __shared__ float rS[256], rS2[256], rSA[256];
  int t = blockIdx.x;
  int tid = threadIdx.x;
  int g = tid >> 4, lane16 = tid & 15;
  int h = g >> 2, ten = g & 3;
  const float* src = (ten == 0) ? fsrt : (ten == 1) ? flng : (ten == 2) ? dlt : vd;
  size_t base = (size_t)t * CD + h * CDV;
  float s = 0.f, s2 = 0.f, sa = 0.f;
  for (int e = 0; e < 16; ++e) {
    float xv = src[base + lane16 * 16 + e];
    s += xv; s2 += xv * xv; sa += fabsf(xv);
  }
  rS[tid] = s; rS2[tid] = s2; rSA[tid] = sa;
  __syncthreads();
  if (tid < 16) {
    float S = 0.f, S2 = 0.f, SA = 0.f;
    for (int e = 0; e < 16; ++e) { S += rS[tid * 16 + e]; S2 += rS2[tid * 16 + e]; SA += rSA[tid * 16 + e]; }
    float mean = S / 256.f;
    float var  = S2 / 256.f - mean * mean;
    float am   = SA / 256.f;
    float l2   = sqrtf(S2);
    int hh = tid >> 2, tt = tid & 3;
    _Float16* gp = gate_in + ((size_t)t * CH + hh) * KPAD + CD + tt * 4;
    gp[0] = (_Float16)mean; gp[1] = (_Float16)var;
    gp[2] = (_Float16)am;   gp[3] = (_Float16)l2;
  }
  for (int e = 0; e < 16; ++e) {   // replicate hidden_states per head
    int idx = tid + e * 256;
    int hh = idx >> 10, c = idx & 1023;
    gate_in[((size_t)t * CH + hh) * KPAD + c] = (_Float16)hs[(size_t)t * CD + c];
  }
  if (tid < 64) {                  // zero pad 1040..1055
    int hh = tid >> 4, c = 1040 + (tid & 15);
    gate_in[((size_t)t * CH + hh) * KPAD + c] = (_Float16)0.f;
  }
}

// ---------------------------------------------------------------- gate MLP tail + combine + RMSNorm
__global__ __launch_bounds__(256)
void combine_kernel(const float* __restrict__ h1, const float* __restrict__ m1b,
                    const float* __restrict__ m2w, const float* __restrict__ m2b,
                    const float* __restrict__ ltemp, const float* __restrict__ crl,
                    const float* __restrict__ nw,
                    const float* __restrict__ fsrt, const float* __restrict__ flng,
                    const float* __restrict__ dlt, const float* __restrict__ vd,
                    _Float16* __restrict__ oh) {
  __shared__ float red[256];
  __shared__ float lg[4];
  __shared__ float ps[5];   // p0..p3, conv residual gate
  int t = blockIdx.x;
  int tid = threadIdx.x;
  float temp = log1pf(__expf(ltemp[0]));   // softplus

  for (int h = 0; h < CH; ++h) {
    size_t row = (size_t)t * CH + h;
    float pl[4] = {0.f, 0.f, 0.f, 0.f};
#pragma unroll
    for (int e = 0; e < 4; ++e) {
      int c = tid + e * 256;
      float x = h1[row * 1024 + c] + m1b[c];
      float g = 0.5f * x * (1.f + tanhf(0.7978845608f * (x + 0.044715f * x * x * x)));
#pragma unroll
      for (int o = 0; o < 4; ++o) pl[o] += g * m2w[o * 1024 + c];
    }
    for (int o = 0; o < 4; ++o) {
      red[tid] = pl[o];
      __syncthreads();
      for (int st = 128; st > 0; st >>= 1) {
        if (tid < st) red[tid] += red[tid + st];
        __syncthreads();
      }
      if (tid == 0) lg[o] = red[0] + m2b[o];
      __syncthreads();
    }
    if (tid == 0) {
      float li[4], ex[4], sum = 0.f;
      float mx = -1e30f;
#pragma unroll
      for (int o = 0; o < 4; ++o) { li[o] = lg[o] / temp; mx = fmaxf(mx, li[o]); }
#pragma unroll
      for (int o = 0; o < 4; ++o) { ex[o] = __expf(li[o] - mx); sum += ex[o]; }
      float p[4], sum2 = 0.f;
#pragma unroll
      for (int o = 0; o < 4; ++o) { p[o] = fmaxf(ex[o] / sum, 0.02f); sum2 += p[o]; }
#pragma unroll
      for (int o = 0; o < 4; ++o) ps[o] = p[o] / sum2;
      ps[4] = 1.f / (1.f + __expf(-crl[h]));
    }
    __syncthreads();
    size_t base = (size_t)t * CD + h * CDV + tid;
    float sh = fsrt[base], ln = flng[base], de = dlt[base], vv = vd[base];
    float o = ps[0] * sh + ps[1] * ln + ps[2] * de + ps[3] * vv + ps[4] * sh;
    red[tid] = o * o;
    __syncthreads();
    for (int st = 128; st > 0; st >>= 1) {
      if (tid < st) red[tid] += red[tid + st];
      __syncthreads();
    }
    float rms = rsqrtf(red[0] / 256.f + 1e-5f);
    oh[base] = (_Float16)(o * rms * nw[tid]);
    __syncthreads();
  }
}

// ---------------------------------------------------------------- launch
extern "C" void kernel_launch(void* const* d_in, const int* in_sizes, int n_in,
                              void* d_out, int out_size, void* d_ws, size_t ws_size,
                              hipStream_t stream) {
  const float* hs    = (const float*)d_in[0];
  const float* q_w   = (const float*)d_in[1];
  const float* k_w   = (const float*)d_in[2];
  const float* v_w   = (const float*)d_in[3];
  const float* b_w   = (const float*)d_in[4];
  const float* q_cw  = (const float*)d_in[5];
  const float* k_cw  = (const float*)d_in[6];
  const float* v_cw  = (const float*)d_in[7];
  const float* fsw   = (const float*)d_in[8];
  const float* flw   = (const float*)d_in[9];
  const float* m1w   = (const float*)d_in[10];
  const float* m1b   = (const float*)d_in[11];
  const float* m2w   = (const float*)d_in[12];
  const float* m2b   = (const float*)d_in[13];
  const float* ltemp = (const float*)d_in[14];
  const float* crl   = (const float*)d_in[15];
  const float* nw    = (const float*)d_in[16];
  const float* o_w   = (const float*)d_in[17];

  char* ws = (char*)d_ws;
  size_t off = 0;
  auto alloc = [&](size_t bytes) {
    size_t cur = off;
    off += (bytes + 255) & ~(size_t)255;
    return cur;
  };

  _Float16* hs_h  = (_Float16*)(ws + alloc((size_t)NTOK * CD * 2));
  _Float16* qw_h  = (_Float16*)(ws + alloc((size_t)CD * CD * 2));
  _Float16* kw_h  = (_Float16*)(ws + alloc((size_t)CD * CD * 2));
  _Float16* vw_h  = (_Float16*)(ws + alloc((size_t)CD * CD * 2));
  _Float16* ow_h  = (_Float16*)(ws + alloc((size_t)CD * CD * 2));
  _Float16* m1w_h = (_Float16*)(ws + alloc((size_t)CD * KPAD * 2));
  float* qp    = (float*)(ws + alloc((size_t)NTOK * CD * 4));
  float* kp    = (float*)(ws + alloc((size_t)NTOK * CD * 4));
  float* vp    = (float*)(ws + alloc((size_t)NTOK * CD * 4));
  float* qc    = (float*)(ws + alloc((size_t)NTOK * CD * 4));
  float* kc    = (float*)(ws + alloc((size_t)NTOK * CD * 4));
  float* vc    = (float*)(ws + alloc((size_t)NTOK * CD * 4));
  float* beta  = (float*)(ws + alloc((size_t)NROWS * 4));
  float* w_buf = (float*)(ws + alloc((size_t)NTOK * CD * 4));
  float* delta = (float*)(ws + alloc((size_t)NTOK * CD * 4));
  float* fshort= (float*)(ws + alloc((size_t)NTOK * CD * 4));
  float* flong = (float*)(ws + alloc((size_t)NTOK * CD * 4));
  _Float16* gate = (_Float16*)(ws + alloc((size_t)NROWS * KPAD * 2));
  float* h1    = (float*)(ws + alloc((size_t)NROWS * 1024 * 4));
  // aliases (inputs dead by the time these are written)
  float* qn = qp;
  float* kn = kp;
  float* u_buf = vp;
  _Float16* oh = (_Float16*)qc;

  const int TPB = 256;
  int nHS = NTOK * CD;
  cvt_f16_kernel<<<(nHS + TPB - 1) / TPB, TPB, 0, stream>>>(hs, hs_h, nHS);
  int nW = CD * CD;
  cvt_f16_kernel<<<(nW + TPB - 1) / TPB, TPB, 0, stream>>>(q_w, qw_h, nW);
  cvt_f16_kernel<<<(nW + TPB - 1) / TPB, TPB, 0, stream>>>(k_w, kw_h, nW);
  cvt_f16_kernel<<<(nW + TPB - 1) / TPB, TPB, 0, stream>>>(v_w, vw_h, nW);
  cvt_f16_kernel<<<(nW + TPB - 1) / TPB, TPB, 0, stream>>>(o_w, ow_h, nW);
  int nM1 = CD * KPAD;
  cvt_pad_kernel<<<(nM1 + TPB - 1) / TPB, TPB, 0, stream>>>(m1w, m1w_h, CD, 1040, KPAD);

  dim3 gProj(CD / GT_N, NTOK / GT_M);   // (8, 64)
  gemm_f16f32_kernel<<<gProj, TPB, 0, stream>>>(hs_h, qw_h, qp, NTOK, CD, CD);
  gemm_f16f32_kernel<<<gProj, TPB, 0, stream>>>(hs_h, kw_h, kp, NTOK, CD, CD);
  gemm_f16f32_kernel<<<gProj, TPB, 0, stream>>>(hs_h, vw_h, vp, NTOK, CD, CD);

  shortconv_silu_kernel<<<nHS / TPB, TPB, 0, stream>>>(qp, q_cw, qc, nHS);
  shortconv_silu_kernel<<<nHS / TPB, TPB, 0, stream>>>(kp, k_cw, kc, nHS);
  shortconv_silu_kernel<<<nHS / TPB, TPB, 0, stream>>>(vp, v_cw, vc, nHS);

  beta_kernel<<<NROWS / TPB, TPB, 0, stream>>>(hs, b_w, beta, NROWS);

  deltaprep_kernel<<<CB * CH * NCHUNK, TPB, 0, stream>>>(qc, kc, vc, beta,
                                                         qn, kn, u_buf, w_buf);

  scan_kernel<<<CB * CH * 16, TPB, 30720 * sizeof(float), stream>>>(
      qn, kn, u_buf, w_buf, delta);

  fir_kernel<<<CB * CH * (CL / 64), TPB, 127 * 256 * sizeof(float), stream>>>(
      vc, fsw, flw, fshort, flong);

  stats_kernel<<<NTOK, TPB, 0, stream>>>(hs, fshort, flong, delta, vc, gate);

  dim3 gMlp(CD / GT_N, NROWS / GT_M);   // (8, 256)
  gemm_f16f32_kernel<<<gMlp, TPB, 0, stream>>>(gate, m1w_h, h1, NROWS, CD, KPAD);

  combine_kernel<<<NTOK, TPB, 0, stream>>>(h1, m1b, m2w, m2b, ltemp, crl, nw,
                                           fshort, flong, delta, vc, oh);

  gemm_f16f32_kernel<<<gProj, TPB, 0, stream>>>(oh, ow_h, (float*)d_out,
                                                NTOK, CD, CD);
}
